// RPEMultiHeadAttention_79551384256915
// MI455X (gfx1250) — matrix operand order repaired and checked by reading, not compiled, not run on hardware
//
#include <hip/hip_runtime.h>

// ---------------------------------------------------------------------------
// RPE multi-head attention for gfx1250 (MI455X), bf16 WMMA everywhere.
// T=S=1024, B=4, DM=1024, H=16, D=64.
// ---------------------------------------------------------------------------

typedef __bf16  bf16_t;
typedef __attribute__((ext_vector_type(16))) __bf16 v16bf;
typedef __attribute__((ext_vector_type(8)))  __bf16 v8bf;
typedef __attribute__((ext_vector_type(8)))  float  v8f;

static __device__ __forceinline__ bf16_t f2bf(float f) {
    unsigned u = __builtin_bit_cast(unsigned, f);
    u = (u + 0x7FFFu + ((u >> 16) & 1u)) >> 16;       // round-to-nearest-even
    unsigned short s = (unsigned short)u;
    return __builtin_bit_cast(bf16_t, s);
}

static __device__ __forceinline__ v8f wmma_bf16(v16bf a, v16bf b, v8f c) {
    return __builtin_amdgcn_wmma_f32_16x16x32_bf16(
        /*neg_a=*/false, a, /*neg_b=*/false, b,
        /*c_mod=*/(short)0, c, /*reuse_a=*/false, /*reuse_b=*/false);
}

// A fragment (16x32 bf16), row-major src. lane: row = lane&15; sel = lane>>4
// selects K-half per the ISA A layout: K = sel*8..+7 and 16+sel*8..+7.
static __device__ __forceinline__ v16bf load_a_bf(const bf16_t* row, int k0, int sel) {
    union { v16bf v; v8bf h[2]; } u;
    u.h[0] = *(const v8bf*)(row + k0 + sel * 8);
    u.h[1] = *(const v8bf*)(row + k0 + 16 + sel * 8);
    return u.v;
}

// B fragment (32x16 bf16) from K-major (i.e. B^T row-major) storage.
// lane: col n = lane&15 row of src; K = sel*8..+7 and 16+sel*8..+7.
static __device__ __forceinline__ v16bf load_b_bf(const bf16_t* row, int k0, int sel) {
    union { v16bf v; v8bf h[2]; } u;
    u.h[0] = *(const v8bf*)(row + k0 + sel * 8);
    u.h[1] = *(const v8bf*)(row + k0 + 16 + sel * 8);
    return u.v;
}

// ---------------------------------------------------------------------------
// Kernel 0: prep (single bandwidth-bound pass, all conversions off the hot
// path).  Regions (element index space):
//   [0        , 3145728 )  Wt[z][n][k]  = W_z[k][n]          (bf16, transposed)
//   [3145728  , 15728640)  Xb[z][m][k]  = X_z[m][k]          (bf16)
//   [15728640 , 15859712)  relB[j][d]   (2048 rows, row 2047 zeroed)
//   [15859712 , 15994880)  T2t[d][j]    (stride 2112, cols >= 2047 zeroed)
// ---------------------------------------------------------------------------
__global__ __launch_bounds__(256) void prep_kernel(
    const float* __restrict__ Wq, const float* __restrict__ Wk,
    const float* __restrict__ Wv,
    const float* __restrict__ Xq, const float* __restrict__ Xk,
    const float* __restrict__ Xv,
    const float* __restrict__ rel, const float* __restrict__ orel,
    bf16_t* __restrict__ Wt, bf16_t* __restrict__ Xb,
    bf16_t* __restrict__ relB, bf16_t* __restrict__ T2t)
{
    int idx = blockIdx.x * 256 + threadIdx.x;
    const bf16_t zero = __builtin_bit_cast(bf16_t, (unsigned short)0);
    if (idx < 3145728) {                         // Wt[z][n][k] = W_z[k][n]
        int z = idx >> 20, r = idx & 1048575, n = r >> 10, k = r & 1023;
        const float* W = (z == 0) ? Wq : (z == 1) ? Wk : Wv;
        Wt[idx] = f2bf(W[k * 1024 + n]);
    } else if (idx < 15728640) {                 // Xb: straight convert
        int i = idx - 3145728;
        int z = i >> 22, r = i & 4194303;
        const float* X = (z == 0) ? Xq : (z == 1) ? Xk : Xv;
        Xb[i] = f2bf(X[r]);
    } else if (idx < 15859712) {                 // relB[j][d]
        int i = idx - 15728640, j = i >> 6, d = i & 63;
        relB[i] = (j < 2047) ? f2bf(rel[j * 64 + d]) : zero;
    } else if (idx < 15994880) {                 // T2t[d][j] (stride 2112)
        int i = idx - 15859712, d = i / 2112, j = i % 2112;
        T2t[i] = (j < 2047) ? f2bf(orel[j * 64 + d]) : zero;
    }
}

// ---------------------------------------------------------------------------
// Kernel 1: projection GEMM  Y = X @ W + b  (M=4096, N=1024, K=1024)
// Pure bf16 WMMA inner loop (A pre-converted).  Each wave computes two
// adjacent 16x16 N-tiles so every A fragment feeds 2 WMMAs.
// Block: 256 threads = 8 waves; tile 16(M) x 256(N); grid (4, 256).
// mode 0: out[bh][t][d]   (Q/K head layout == B^T for the score GEMM)
// mode 1: out[bh][d][s]   (V transposed  == B^T for the output GEMM)
// ---------------------------------------------------------------------------
__global__ __launch_bounds__(256) void proj_kernel(
    const bf16_t* __restrict__ Xb, const bf16_t* __restrict__ Wt,
    const float* __restrict__ bias, bf16_t* __restrict__ Out, int mode)
{
    int tid = threadIdx.x;
    int wave = tid >> 5, lane = tid & 31;
    int n16 = lane & 15, sel = lane >> 4;
    int m0 = blockIdx.y * 16;
    int nb = blockIdx.x * 256 + wave * 32;

    const bf16_t* arow  = Xb + (size_t)(m0 + n16) * 1024;
    const bf16_t* brow0 = Wt + (size_t)(nb + n16) * 1024;
    const bf16_t* brow1 = brow0 + 16 * 1024;

    v8f c0 = {}, c1 = {};
#pragma unroll 4
    for (int k0 = 0; k0 < 1024; k0 += 32) {
        __builtin_prefetch((const void*)(brow0 + k0 + 512), 0, 1);
        __builtin_prefetch((const void*)(brow1 + k0 + 512), 0, 1);
        v16bf a = load_a_bf(arow, k0, sel);
        c0 = wmma_bf16(a, load_b_bf(brow0, k0, sel), c0);
        c1 = wmma_bf16(a, load_b_bf(brow1, k0, sel), c1);
    }

#pragma unroll
    for (int half = 0; half < 2; ++half) {
        int n = nb + half * 16 + n16;
        float bn = bias[n];
        int h = n >> 6, d = n & 63;
        v8f c = half ? c1 : c0;
#pragma unroll
        for (int j = 0; j < 8; ++j) {
            int m = m0 + j + sel * 8;
            int t = m >> 2, b = m & 3;                   // row m = t*B + b
            size_t addr = (mode == 0)
                ? (((size_t)(b * 16 + h)) * 1024 + t) * 64 + d    // [bh][t][d]
                : (((size_t)(b * 16 + h)) * 64 + d) * 1024 + t;   // [bh][d][s]
            Out[addr] = f2bf(c[j] + bn);
        }
    }
}

// ---------------------------------------------------------------------------
// Kernel 2: attention. One block = one (bh, 16-row t-tile).
// Dynamic LDS (129 KB):
//   [0      , 66560)  P   fp32 [16][1040]   (reused later: Wn bf16 [16][1024]
//                                            at 0, Wsh bf16 [16][1056] at 32768)
//   [66560  ,132096)  Sc  fp32 [16][1024]   (reused: 8 partial out tiles)
// ---------------------------------------------------------------------------
__global__ __launch_bounds__(256) void attn_kernel(
    const bf16_t* __restrict__ Qh, const bf16_t* __restrict__ Kh,
    const bf16_t* __restrict__ Vt, const bf16_t* __restrict__ relB,
    const bf16_t* __restrict__ T2t, float* __restrict__ Out)
{
    extern __shared__ char smem[];
    float*  P      = (float*)smem;                 // [16][1040]
    float*  Sc     = (float*)(smem + 66560);       // [16][1024]
    bf16_t* Wn     = (bf16_t*)smem;                // [16][1024]
    bf16_t* Wsh    = (bf16_t*)(smem + 32768);      // [16][1056]
    float*  outbuf = Sc;                           // [8][16][16]

    int tid = threadIdx.x;
    int wave = tid >> 5, lane = tid & 31;
    int n16 = lane & 15, sel = lane >> 4;
    int t0 = blockIdx.x * 16;
    int bh = blockIdx.y;
    int jbase = 1008 - t0;                          // j = c + jbase

    // Q A-fragments (shared by all stages of this wave): rows t0..t0+15, D=64.
    const bf16_t* qrow = Qh + ((size_t)bh * 1024 + t0 + n16) * 64;
    v16bf aq0 = load_a_bf(qrow, 0, sel);
    v16bf aq1 = load_a_bf(qrow, 32, sel);

    // ---- Stage 1: P[t][c] = q[t] . table[c + jbase]  (65 N-tiles) ----------
    for (int jt = wave; jt < 65; jt += 8) {
        const bf16_t* brow = relB + (size_t)(jbase + jt * 16 + n16) * 64;
        v8f c = {};
        c = wmma_bf16(aq0, load_b_bf(brow, 0, sel), c);
        c = wmma_bf16(aq1, load_b_bf(brow, 32, sel), c);
        int col = jt * 16 + n16;
#pragma unroll
        for (int j = 0; j < 8; ++j) P[(j + sel * 8) * 1040 + col] = c[j];
    }
    __syncthreads();

    // ---- Stage 2: scores = (Q K^T + gather(P)) * scale  (64 N-tiles) -------
    const bf16_t* Kbase = Kh + (size_t)bh * 1024 * 64;
    for (int st = wave; st < 64; st += 8) {
        const bf16_t* brow = Kbase + (size_t)(st * 16 + n16) * 64;
        v8f c = {};
        c = wmma_bf16(aq0, load_b_bf(brow, 0, sel), c);
        c = wmma_bf16(aq1, load_b_bf(brow, 32, sel), c);
        int s = st * 16 + n16;
#pragma unroll
        for (int j = 0; j < 8; ++j) {
            int m = j + sel * 8;
            Sc[m * 1024 + s] = (c[j] + P[m * 1040 + (s - m + 15)]) * 0.125f;
        }
    }
    __syncthreads();

    // ---- Stage 3a: zero the Wn/Wsh region (66560 B = 16640 dwords) ---------
    for (int i = tid; i < 16640; i += 256) ((int*)smem)[i] = 0;
    __syncthreads();

    // ---- Stage 3b: row softmax (16 lanes per row via shfl_xor width 16) ----
    {
        int row = tid >> 4, l16 = tid & 15;
        const float* sr = Sc + row * 1024;
        float mx = -3.4e38f;
        for (int s = l16; s < 1024; s += 16) mx = fmaxf(mx, sr[s]);
        for (int o = 8; o; o >>= 1) mx = fmaxf(mx, __shfl_xor(mx, o, 16));
        float sum = 0.f;
        for (int s = l16; s < 1024; s += 16) sum += __expf(sr[s] - mx);
        for (int o = 8; o; o >>= 1) sum += __shfl_xor(sum, o, 16);
        float inv = 1.0f / sum;
        for (int s = l16; s < 1024; s += 16) {
            bf16_t wb = f2bf(__expf(sr[s] - mx) * inv);
            Wn[row * 1024 + s] = wb;                   // plain weights
            Wsh[row * 1056 + (s - row + 15)] = wb;     // shifted weights
        }
    }
    __syncthreads();

    // ---- Stage 4/5: out = Wn @ V^T  (waves 0-3)  +  Wsh @ table2^T (4-7) ---
    v8f co = {};
    if (wave < 4) {
        int dt = wave;
        const bf16_t* vb   = Vt + ((size_t)bh * 64 + dt * 16 + n16) * 1024;
        const bf16_t* wrow = Wn + n16 * 1024;
#pragma unroll 4
        for (int k0 = 0; k0 < 1024; k0 += 32)
            co = wmma_bf16(load_a_bf(wrow, k0, sel), load_b_bf(vb, k0, sel), co);
    } else {
        int dt = wave - 4;
        const bf16_t* tb   = T2t + (size_t)(dt * 16 + n16) * 2112 + jbase;
        const bf16_t* wrow = Wsh + n16 * 1056;
        for (int k0 = 0; k0 < 1056; k0 += 32)
            co = wmma_bf16(load_a_bf(wrow, k0, sel), load_b_bf(tb, k0, sel), co);
    }
#pragma unroll
    for (int j = 0; j < 8; ++j)
        outbuf[wave * 256 + (j + sel * 8) * 16 + n16] = co[j];
    __syncthreads();

    // ---- Final: sum content + rel partials, write (T,B,DM) fp32 ------------
    int b = bh >> 4, h = bh & 15;
    for (int i = tid; i < 1024; i += 256) {
        int ml = i >> 6, d = i & 63;
        float v = outbuf[(d >> 4) * 256 + ml * 16 + (d & 15)]
                + outbuf[(4 + (d >> 4)) * 256 + ml * 16 + (d & 15)];
        Out[((size_t)(t0 + ml) * 4 + b) * 1024 + h * 64 + d] = v;
    }
}

// ---------------------------------------------------------------------------
extern "C" void kernel_launch(void* const* d_in, const int* in_sizes, int n_in,
                              void* d_out, int out_size, void* d_ws, size_t ws_size,
                              hipStream_t stream) {
    (void)in_sizes; (void)n_in; (void)out_size; (void)ws_size;
    const float* query = (const float*)d_in[0];
    const float* key_  = (const float*)d_in[1];
    const float* value = (const float*)d_in[2];
    const float* Wq    = (const float*)d_in[3];
    const float* bq    = (const float*)d_in[4];
    const float* Wk    = (const float*)d_in[5];
    const float* bk    = (const float*)d_in[6];
    const float* Wv    = (const float*)d_in[7];
    const float* bv    = (const float*)d_in[8];
    const float* rel   = (const float*)d_in[9];
    const float* orel  = (const float*)d_in[10];
    float* out = (float*)d_out;

    char* ws = (char*)d_ws;
    bf16_t* Wt   = (bf16_t*)(ws);                    // [3][1024][1024]  6 MB
    bf16_t* Xb   = (bf16_t*)(ws + 6291456);          // [3][4096][1024] 24 MB
    bf16_t* Qh   = (bf16_t*)(ws + 31457280);         // [64][1024][64]   8 MB
    bf16_t* Khd  = (bf16_t*)(ws + 39845888);         // [64][1024][64]   8 MB
    bf16_t* Vt   = (bf16_t*)(ws + 48234496);         // [64][64][1024]   8 MB
    bf16_t* relB = (bf16_t*)(ws + 56623104);         // [2048][64]
    bf16_t* T2t  = (bf16_t*)(ws + 56885248);         // [64][2112]

    prep_kernel<<<62480, 256, 0, stream>>>(Wq, Wk, Wv, query, key_, value,
                                           rel, orel, Wt, Xb, relB, T2t);
    proj_kernel<<<dim3(4, 256), 256, 0, stream>>>(Xb,           Wt,           bq, Qh,  0);
    proj_kernel<<<dim3(4, 256), 256, 0, stream>>>(Xb + 4194304, Wt + 1048576, bk, Khd, 0);
    proj_kernel<<<dim3(4, 256), 256, 0, stream>>>(Xb + 8388608, Wt + 2097152, bv, Vt,  1);
    attn_kernel<<<dim3(64, 64), 256, 132096, stream>>>(Qh, Khd, Vt, relB, T2t, out);
}
